// LSTM_61735859913401
// MI455X (gfx1250) — matrix operand order repaired and checked by reading, not compiled, not run on hardware
//
#include <hip/hip_runtime.h>

// ---------------------------------------------------------------------------
// Persistent LSTM for MI455X (gfx1250).
//   * w_hh (8192x2048) repacked to bf16 WMMA A-fragments, resident in LDS
//     (256 KB per WGP, 128 WGPs hold all 32 MB) -- avoids 262 TB of HBM
//     traffic (11 s at 23.3 TB/s) that a streaming design would need.
//   * TDM (tensor_load_to_lds) stages the weight slice into LDS once.
//   * Per step: v_wmma_f32_16x16x32_bf16 GEMV, device-wide barrier via
//     agent-scope atomics, h republished through L2.
// ---------------------------------------------------------------------------

#define H        2048
#define T_STEPS  4096
#define NWG      128          // workgroups; each owns SLICE hidden units
#define SLICE    16
#define BLOCK    256          // 8 wave32 waves: 2 per gate (k-split)
#define KK_TOT   64           // 2048 / 32 k-chunks
#define WPACK_BYTES_PER_WG (4 * KK_TOT * 32 * 32)   // 262144 B per WG
#define WPACK_TOTAL ((size_t)NWG * WPACK_BYTES_PER_WG) // 32 MB

#define SMEM_W   262144
#define SMEM_H   (SMEM_W + 4096)          // not an offset: see layout below
#define SMEM_BYTES (262144 /*W*/ + 4096 /*h bf16*/ + 512 /*z*/ + 64 /*y*/)

typedef __attribute__((ext_vector_type(16))) __bf16  v16bf;
typedef __attribute__((ext_vector_type(8)))  float   v8f;
typedef __attribute__((ext_vector_type(4)))  unsigned int u32x4;
typedef __attribute__((ext_vector_type(8)))  int     i32x8;
typedef __attribute__((ext_vector_type(4)))  int     i32x4;

__device__ __forceinline__ unsigned short f2bf(float f) {
  unsigned u = __float_as_uint(f);
  u += 0x7FFFu + ((u >> 16) & 1u);          // round-to-nearest-even
  return (unsigned short)(u >> 16);
}
__device__ __forceinline__ float sigmoid_fast(float x) {
  return 1.0f / (1.0f + __expf(-x));
}
__device__ __forceinline__ float tanh_fast(float x) {
  return 1.0f - 2.0f / (__expf(2.0f * x) + 1.0f);
}

// ---------------------------------------------------------------------------
// Repack w_hh (fp32 row-major [4H][H]) into bf16 WMMA A-fragment order:
//   Wpack[wg][gate][kk][lane][16 halves], 32 B per lane, so the persistent
//   kernel loads each A fragment as two contiguous ds_load_b128.
// A order used here (contiguous k; B at line 155 matches): lanes 0-15 hold K={k0..k0+15} of row M,
//                        lanes 16-31 hold K={k0+16..k0+31}.
// ---------------------------------------------------------------------------
__global__ void repack_w(const float* __restrict__ w_hh,
                         unsigned short* __restrict__ Wpack) {
  int gid  = blockIdx.x * 256 + threadIdx.x;       // 0 .. 1048575
  int lane = gid & 31;
  int kk   = (gid >> 5) & 63;
  int q    = (gid >> 11) & 3;
  int wg   = gid >> 13;
  int r    = lane & 15;
  int k0   = kk * 32;
  int kadd = (lane < 16) ? 0 : 16;
  size_t R = (size_t)q * H + (size_t)wg * SLICE + r;
  const float* src = w_hh + R * H;
  unsigned short* dst = Wpack + (size_t)gid * 16;
#pragma unroll
  for (int hh = 0; hh < 16; ++hh) {
    int K = k0 + kadd + hh;
    dst[hh] = f2bf(src[K]);
  }
}

// ---------------------------------------------------------------------------
// Persistent kernel: 128 blocks x 256 threads, one block per WGP.
// ---------------------------------------------------------------------------
__launch_bounds__(BLOCK, 1)
__global__ void lstm_persistent(const float* __restrict__ inputs,
                                const float* __restrict__ w_ih,
                                const float* __restrict__ b_ih,
                                const float* __restrict__ b_hh,
                                const float* __restrict__ w_out,
                                const unsigned short* __restrict__ Wpack,
                                float* __restrict__ hbufg,
                                unsigned int* __restrict__ syncw,
                                float* __restrict__ ypart,
                                float* __restrict__ out) {
  extern __shared__ __align__(128) char smem[];
  unsigned short* sW = (unsigned short*)smem;                   // 256 KB weights
  unsigned short* sH = (unsigned short*)(smem + 262144);        // 4 KB h (bf16)
  float* sZ = (float*)(smem + 262144 + 4096);                   // 8 waves x 16
  float* sY = (float*)(smem + 262144 + 4096 + 512);             // 16 partials

  const int tid  = threadIdx.x;
  const int wg   = blockIdx.x;
  const int base = wg * SLICE;
  const int wave = tid >> 5;
  const int lane = tid & 31;
  const int gate = wave >> 1;
  const int half = wave & 1;

  // ---- stage this WGP's weight slice into LDS via the Tensor Data Mover ----
  if (wave == 0) {
    unsigned long long ga = (unsigned long long)(uintptr_t)Wpack
                          + (unsigned long long)wg * (unsigned long long)WPACK_BYTES_PER_WG;
    unsigned lds_addr = (unsigned)(uintptr_t)sW;   // low 32 bits = LDS byte offset
    u32x4 g0 = {};
    g0[0] = 1u;                                          // count = 1 descriptor
    g0[1] = lds_addr;                                    // lds_addr
    g0[2] = (unsigned)(ga & 0xFFFFFFFFu);                // global_addr[31:0]
    g0[3] = (unsigned)((ga >> 32) & 0x01FFFFFFu) | (2u << 30);  // addr hi | type=2
    i32x8 g1 = {};
    g1[0] = (int)(1u << 16);       // data_size = 1 -> 2 bytes/elem
    g1[1] = (int)0x80000000u;      // tensor_dim0 = 32768 (lo16 at [31:16])
    g1[2] = (int)0x00040000u;      // tensor_dim0 hi = 0 | tensor_dim1 = 4
    g1[3] = (int)0x80000000u;      // tile_dim0 = 32768
    g1[4] = 4;                     // tile_dim1 = 4, tile_dim2 = 0
    g1[5] = 32768;                 // tensor_dim0_stride = 32768 elems
    g1[6] = 0;
    g1[7] = 0;
    i32x4 g2 = {};
    i32x4 g3 = {};
#if defined(__clang_major__) && (__clang_major__ >= 23)
    i32x8 g1b = {};
    __builtin_amdgcn_tensor_load_to_lds(g0, g1, g2, g3, g1b, 0);
#else
    __builtin_amdgcn_tensor_load_to_lds(g0, g1, g2, g3, 0);
#endif
    __builtin_amdgcn_s_wait_tensorcnt(0);
  }

  // h(-1) = 0
  for (int e = tid; e < H; e += BLOCK) sH[e] = 0;

  // per-hidden-unit constants for the gate-combine threads (tid 0..15)
  float wx0 = 0, wx1 = 0, wx2 = 0, wx3 = 0;
  float bb0 = 0, bb1 = 0, bb2 = 0, bb3 = 0;
  float wo = 0, c = 0.0f;
  if (tid < SLICE) {
    int j = base + tid;
    wx0 = w_ih[0 * H + j]; wx1 = w_ih[1 * H + j];
    wx2 = w_ih[2 * H + j]; wx3 = w_ih[3 * H + j];
    bb0 = b_ih[0 * H + j] + b_hh[0 * H + j];
    bb1 = b_ih[1 * H + j] + b_hh[1 * H + j];
    bb2 = b_ih[2 * H + j] + b_hh[2 * H + j];
    bb3 = b_ih[3 * H + j] + b_hh[3 * H + j];
    wo  = w_out[j];
  }
  __syncthreads();

  const int kkbase = half * 32;
  const v16bf* aw = (const v16bf*)sW + (size_t)gate * KK_TOT * 32 + lane;

  for (int t = 0; t < T_STEPS; ++t) {
    // ---- z-slice = W_gate[16 x 2048] @ h, via WMMA bf16 ----
    v8f a0 = {0.f,0.f,0.f,0.f,0.f,0.f,0.f,0.f};
    v8f a1 = a0, a2 = a0, a3 = a0;
    const unsigned short* hp = sH + ((lane & 16) ? 16 : 0);
#pragma unroll
    for (int i = 0; i < 32; i += 4) {
      int kk = kkbase + i;
      v16bf A0 = aw[(size_t)(kk + 0) * 32];
      v16bf B0 = *(const v16bf*)(hp + (kk + 0) * 32);
      a0 = __builtin_amdgcn_wmma_f32_16x16x32_bf16(false, A0, false, B0,
                                                   (short)0, a0, false, false);
      v16bf A1 = aw[(size_t)(kk + 1) * 32];
      v16bf B1 = *(const v16bf*)(hp + (kk + 1) * 32);
      a1 = __builtin_amdgcn_wmma_f32_16x16x32_bf16(false, A1, false, B1,
                                                   (short)0, a1, false, false);
      v16bf A2 = aw[(size_t)(kk + 2) * 32];
      v16bf B2 = *(const v16bf*)(hp + (kk + 2) * 32);
      a2 = __builtin_amdgcn_wmma_f32_16x16x32_bf16(false, A2, false, B2,
                                                   (short)0, a2, false, false);
      v16bf A3 = aw[(size_t)(kk + 3) * 32];
      v16bf B3 = *(const v16bf*)(hp + (kk + 3) * 32);
      a3 = __builtin_amdgcn_wmma_f32_16x16x32_bf16(false, A3, false, B3,
                                                   (short)0, a3, false, false);
    }
    v8f acc = (a0 + a1) + (a2 + a3);
    // every C column equals W.h; lane l holds rows (l>=16 ? 8..15 : 0..7)
    if ((lane & 15) == 0) {
      int off = wave * 16 + ((lane >> 4) << 3);
#pragma unroll
      for (int v = 0; v < 8; ++v) sZ[off + v] = acc[v];
    }
    __syncthreads();

    // ---- gate combine: thread j owns hidden unit base+j ----
    if (tid < SLICE) {
      float x = inputs[t];
      float zi = sZ[0 * 16 + tid] + sZ[1 * 16 + tid] + x * wx0 + bb0;
      float zf = sZ[2 * 16 + tid] + sZ[3 * 16 + tid] + x * wx1 + bb1;
      float zg = sZ[4 * 16 + tid] + sZ[5 * 16 + tid] + x * wx2 + bb2;
      float zo = sZ[6 * 16 + tid] + sZ[7 * 16 + tid] + x * wx3 + bb3;
      float gi = sigmoid_fast(zi);
      float gf = sigmoid_fast(zf);
      float gg = tanh_fast(zg);
      float go = sigmoid_fast(zo);
      c = gf * c + gi * gg;
      float hv = go * tanh_fast(c);
      __hip_atomic_store(&hbufg[base + tid], hv, __ATOMIC_RELAXED,
                         __HIP_MEMORY_SCOPE_AGENT);
      sY[tid] = wo * hv;
      if (t == T_STEPS - 1) {
        out[T_STEPS + base + tid]     = hv;   // final h
        out[T_STEPS + H + base + tid] = c;    // final c
      }
    }
    __syncthreads();

    // ---- y partial + device-wide step barrier ----
    if (tid == 0) {
      float y = 0.f;
#pragma unroll
      for (int j2 = 0; j2 < SLICE; ++j2) y += sY[j2];
      ypart[(size_t)t * NWG + wg] = y;
      unsigned old = __hip_atomic_fetch_add(&syncw[0], 1u, __ATOMIC_ACQ_REL,
                                            __HIP_MEMORY_SCOPE_AGENT);
      unsigned tgt = (unsigned)(t + 1);
      if (old + 1u == (unsigned)NWG * tgt) {
        __hip_atomic_store(&syncw[1], tgt, __ATOMIC_RELEASE,
                           __HIP_MEMORY_SCOPE_AGENT);
      } else {
        while (__hip_atomic_load(&syncw[1], __ATOMIC_ACQUIRE,
                                 __HIP_MEMORY_SCOPE_AGENT) < tgt)
          __builtin_amdgcn_s_sleep(2);
      }
    }
    __syncthreads();

    // ---- pull the freshly published h, convert to bf16 in LDS ----
    if (t < T_STEPS - 1) {
#pragma unroll
      for (int e = 0; e < H / BLOCK; ++e) {
        int idx = tid * (H / BLOCK) + e;
        float hv = __hip_atomic_load(&hbufg[idx], __ATOMIC_RELAXED,
                                     __HIP_MEMORY_SCOPE_AGENT);
        sH[idx] = f2bf(hv);
      }
      __syncthreads();
    }
  }
}

// ---------------------------------------------------------------------------
// outputs[t] = sum over 128 per-WG partials + b_out
// ---------------------------------------------------------------------------
__global__ void reduce_y(const float* __restrict__ ypart,
                         const float* __restrict__ b_out,
                         float* __restrict__ out) {
  int t = blockIdx.x * 256 + threadIdx.x;
  if (t < T_STEPS) {
    float s = b_out[0];
    for (int w = 0; w < NWG; ++w) s += ypart[(size_t)t * NWG + w];
    out[t] = s;
  }
}

extern "C" void kernel_launch(void* const* d_in, const int* in_sizes, int n_in,
                              void* d_out, int out_size, void* d_ws, size_t ws_size,
                              hipStream_t stream) {
  (void)in_sizes; (void)n_in; (void)out_size; (void)ws_size;
  const float* inputs = (const float*)d_in[0];
  const float* w_ih   = (const float*)d_in[1];
  const float* w_hh   = (const float*)d_in[2];
  const float* b_ih   = (const float*)d_in[3];
  const float* b_hh   = (const float*)d_in[4];
  const float* w_out  = (const float*)d_in[5];
  const float* b_out  = (const float*)d_in[6];
  float* out = (float*)d_out;

  char* ws = (char*)d_ws;
  unsigned short* Wpack = (unsigned short*)ws;                       // 32 MB
  float*        hbufg   = (float*)(ws + WPACK_TOTAL);                // 8 KB
  unsigned int* syncw   = (unsigned int*)(ws + WPACK_TOTAL + 8192);  // 256 B
  float*        ypart   = (float*)(ws + WPACK_TOTAL + 8192 + 256);   // 2 MB

  // barrier counter/flag must start at 0 every launch (graph-replay safe)
  hipMemsetAsync(syncw, 0, 256, stream);

  repack_w<<<4096, 256, 0, stream>>>(w_hh, Wpack);

  hipFuncSetAttribute((const void*)lstm_persistent,
                      hipFuncAttributeMaxDynamicSharedMemorySize, SMEM_BYTES);
  lstm_persistent<<<NWG, BLOCK, SMEM_BYTES, stream>>>(
      inputs, w_ih, b_ih, b_hh, w_out, Wpack, hbufg, syncw, ypart, out);

  reduce_y<<<T_STEPS / 256, 256, 0, stream>>>(ypart, b_out, out);
}
